// AssocModel_16234976379289
// MI455X (gfx1250) — matrix operand order repaired and checked by reading, not compiled, not run on hardware
//
#include <hip/hip_runtime.h>
#include <hip/hip_bf16.h>

typedef __attribute__((ext_vector_type(16))) _Float16 v16h;
typedef __attribute__((ext_vector_type(8)))  float    v8f;

#define NM 1024
#define NT 512
#define NN 2
#define H  128

// ---------------------------------------------------------------------------
// Kernel 0: pre-pack e_W2 (128x128 f32, row-major [K][N]) into a contiguous-k
// B-fragment layout for v_wmma_f32_16x16x32_f16:
//   fragment (kt, nt): lane L holds column N = nt*16 + (L&15),
//                      half i -> K = kt*32 + (L>>4)*16 + i
// Packed as bpack[((kt*8+nt)*32 + lane)*16 + i] so the fused kernel reads
// one contiguous 32B v16h per lane.
// ---------------------------------------------------------------------------
__global__ void pack_w2_kernel(const float* __restrict__ w2,
                               _Float16* __restrict__ bpack) {
    int tid = blockIdx.x * blockDim.x + threadIdx.x;   // 0..1023
    if (tid >= 32 * 32) return;
    int lane = tid & 31;
    int frag = tid >> 5;            // kt*8 + nt
    int kt = frag >> 3, nt = frag & 7;
    int kbase = kt * 32 + ((lane >> 4) << 4);
    int ncol  = nt * 16 + (lane & 15);
    _Float16* dst = bpack + (size_t)tid * 16;
#pragma unroll
    for (int i = 0; i < 16; ++i)
        dst[i] = (_Float16)w2[(kbase + i) * H + ncol];
}

// ---------------------------------------------------------------------------
// Kernel 1: encoders + projection through e_W1.
// rows 0..1023   : a[m]   = relu(relu(curr@ecW1+b1)@ecW2+b2) @ Wi   (Wi = e_W1[0:128])
// rows 1024..2047: b[n,t] = relu(relu(fut @efW1+b1)@efW2+b2) @ Wj   (Wj = e_W1[128:256])
// One block (128 threads) per row; tiny fraction of total FLOPs.
// ---------------------------------------------------------------------------
__global__ __launch_bounds__(128)
void encode_kernel(const float* __restrict__ curr, const float* __restrict__ fut,
                   const float* __restrict__ ecW1, const float* __restrict__ ecb1,
                   const float* __restrict__ ecW2, const float* __restrict__ ecb2,
                   const float* __restrict__ efW1, const float* __restrict__ efb1,
                   const float* __restrict__ efW2, const float* __restrict__ efb2,
                   const float* __restrict__ eW1,
                   float* __restrict__ aout, float* __restrict__ bout) {
    __shared__ float xs[16];
    __shared__ float l1s[H];
    __shared__ float hs[H];

    const int row = blockIdx.x;
    const int j   = threadIdx.x;
    const bool isFut = row >= NM;
    const int r = isFut ? row - NM : row;

    const float* x  = (isFut ? fut : curr) + (size_t)r * 16;
    const float* W1 = isFut ? efW1 : ecW1;
    const float* b1 = isFut ? efb1 : ecb1;
    const float* W2 = isFut ? efW2 : ecW2;
    const float* b2 = isFut ? efb2 : ecb2;

    if (j < 16) xs[j] = x[j];
    __syncthreads();

    float acc = b1[j];
#pragma unroll
    for (int k = 0; k < 16; ++k) acc = fmaf(xs[k], W1[k * H + j], acc);
    l1s[j] = fmaxf(acc, 0.f);
    __syncthreads();

    acc = b2[j];
    for (int k = 0; k < H; ++k) acc = fmaf(l1s[k], W2[k * H + j], acc);
    hs[j] = fmaxf(acc, 0.f);
    __syncthreads();

    const float* We = eW1 + (isFut ? H * H : 0);   // Wi or Wj block of e_W1
    acc = 0.f;
    for (int k = 0; k < H; ++k) acc = fmaf(hs[k], We[k * H + j], acc);
    (isFut ? bout : aout)[(size_t)r * H + j] = acc;
}

// ---------------------------------------------------------------------------
// Kernel 2: masked least-squares velocity fit + x_pred[n][m][c].
// ---------------------------------------------------------------------------
__global__ void xpred_kernel(const float* __restrict__ curr,
                             const float* __restrict__ hist_xy,
                             const float* __restrict__ hist_mask,
                             float* __restrict__ xpred /*[2][NM][2]*/) {
    int m = blockIdx.x * blockDim.x + threadIdx.x;
    if (m >= NM) return;
    const int K = 10;
    float msum = 0.f, tm = 0.f, ym0 = 0.f, ym1 = 0.f;
#pragma unroll
    for (int k = 0; k < K; ++k) {
        float mk = hist_mask[m * K + k];
        float tk = (float)(k - (K - 1));
        msum += mk; tm += tk * mk;
        ym0 += hist_xy[(m * K + k) * 2 + 0] * mk;
        ym1 += hist_xy[(m * K + k) * 2 + 1] * mk;
    }
    msum = fmaxf(msum, 1.f);
    tm /= msum; ym0 /= msum; ym1 /= msum;
    float num0 = 0.f, num1 = 0.f, den = 0.f;
#pragma unroll
    for (int k = 0; k < K; ++k) {
        float mk = hist_mask[m * K + k];
        float tc = ((float)(k - (K - 1)) - tm) * mk;
        num0 += tc * (hist_xy[(m * K + k) * 2 + 0] - ym0) * mk;
        num1 += tc * (hist_xy[(m * K + k) * 2 + 1] - ym1) * mk;
        den  += tc * tc;
    }
    den = fmaxf(den, 1e-8f);
    float v0 = num0 / den, v1 = num1 / den;
    float x0 = curr[m * 16 + 0], x1 = curr[m * 16 + 1];
#pragma unroll
    for (int n = 0; n < NN; ++n) {
        xpred[(n * NM + m) * 2 + 0] = fmaf((float)(n + 1), v0, x0);
        xpred[(n * NM + m) * 2 + 1] = fmaf((float)(n + 1), v1, x1);
    }
}

// ---------------------------------------------------------------------------
// Kernel 3: fused edge MLP. grid = (tchunk=4, m=1024, n=2), 256 thr = 8 waves.
// Wave w owns 16 rows = t in [tchunk*128 + w*16, +16) at fixed (n,m).
// h1 built in registers directly in f16 A-fragment layout; 32 WMMAs produce
// h2 (never stored); epilogue fuses +b2, relu, dot(W3), shfl_xor reduction.
// ---------------------------------------------------------------------------
__global__ __launch_bounds__(256)
void assoc_fused_kernel(const float* __restrict__ aglob,   // [NM][H]
                        const float* __restrict__ bglob,   // [NN*NT][H]
                        const float* __restrict__ fut,     // [NN][NT][16]
                        const float* __restrict__ xpred,   // [NN][NM][2]
                        const _Float16* __restrict__ bpack,// [32][32][16]
                        const float* __restrict__ eW1,     // rows 256..258 = Wx,Wy,Wt
                        const float* __restrict__ eb1,
                        const float* __restrict__ eb2,
                        const float* __restrict__ eW3,
                        const float* __restrict__ eb3,
                        const float* __restrict__ dummy,
                        float* __restrict__ out)           // [NN][NM][NT+1]
{
    __shared__ _Float16 sB[32 * 32 * 16];  // packed W2 fragments, 32KB
    __shared__ float sA[H];
    __shared__ float sWx[H], sWy[H], sWt[H], sB1[H], sB2[H], sW3[H];

    const int tid  = threadIdx.x;
    const int lane = tid & 31;
    const int wave = tid >> 5;
    const int tchunk = blockIdx.x;
    const int m = blockIdx.y;
    const int n = blockIdx.z;

    // stage packed W2 (L2-resident 32KB) into LDS
    {
        const uint4* src = (const uint4*)bpack;
        uint4* dst = (uint4*)sB;
        for (int i = tid; i < 2048; i += 256) dst[i] = src[i];
    }
    if (tid < H) {
        sA[tid]  = aglob[(size_t)m * H + tid];
        sWx[tid] = eW1[(2 * H + 0) * H + tid];
        sWy[tid] = eW1[(2 * H + 1) * H + tid];
        sWt[tid] = eW1[(2 * H + 2) * H + tid];
        sB1[tid] = eb1[tid];
        sB2[tid] = eb2[tid];
        sW3[tid] = eW3[tid];
    }
    __syncthreads();

    const int khalf = lane >> 4;
    const int nl    = lane & 15;
    const int t0 = tchunk * 128 + wave * 16;
    const int t  = t0 + nl;                 // this lane's GEMM row

    const float xp0 = xpred[(n * NM + m) * 2 + 0];
    const float xp1 = xpred[(n * NM + m) * 2 + 1];
    const float dx = fut[((size_t)(n * NT + t)) * 16 + 0] - xp0;
    const float dy = fut[((size_t)(n * NT + t)) * 16 + 1] - xp1;
    const float taun = (float)(n + 1) * (1.f / (float)NN);

    const float* brow = bglob + (size_t)(n * NT + t) * H;

    // h1 in A-fragment layout: half i of frag kt holds
    //   K = kt*32 + khalf*16 + i   (matches packed B layout)
    v16h af[4];
#pragma unroll
    for (int kt = 0; kt < 4; ++kt) {
#pragma unroll
        for (int i = 0; i < 16; ++i) {
            int k = kt * 32 + (khalf << 4) + i;
            float v = sA[k] + brow[k] + dx * sWx[k] + dy * sWy[k]
                    + taun * sWt[k] + sB1[k];
            af[kt][i] = (_Float16)fmaxf(v, 0.f);
        }
    }

    float lp[8];
#pragma unroll
    for (int r = 0; r < 8; ++r) lp[r] = 0.f;

#pragma unroll
    for (int nt = 0; nt < 8; ++nt) {
        v8f c = {};
#pragma unroll
        for (int kt = 0; kt < 4; ++kt) {
            const v16h bf = *(const v16h*)&sB[(((kt * 8 + nt) * 32) + lane) * 16];
            c = __builtin_amdgcn_wmma_f32_16x16x32_f16(
                    false, af[kt], false, bf, (short)0, c, false, false);
        }
        // C layout: VGPR r -> row M = khalf*8 + r, col N = nt*16 + nl
        const int col = nt * 16 + nl;
        const float bb = sB2[col];
        const float w3 = sW3[col];
#pragma unroll
        for (int r = 0; r < 8; ++r)
            lp[r] = fmaf(fmaxf(c[r] + bb, 0.f), w3, lp[r]);
    }

    // reduce over the 16 lanes of each half (xor masks keep bit4 fixed)
#pragma unroll
    for (int r = 0; r < 8; ++r) {
        float v = lp[r];
        v += __shfl_xor(v, 1, 32);
        v += __shfl_xor(v, 2, 32);
        v += __shfl_xor(v, 4, 32);
        v += __shfl_xor(v, 8, 32);
        lp[r] = v;
    }

    if (nl == 0) {
        const float bias3 = eb3[0];
        float* orow = out + ((size_t)(n * NM + m)) * (NT + 1);
#pragma unroll
        for (int r = 0; r < 8; ++r)
            orow[t0 + khalf * 8 + r] = lp[r] + bias3;   // rows M = khalf*8+r
    }
    if (tchunk == 0 && tid == 0)
        out[((size_t)(n * NM + m)) * (NT + 1) + NT] = dummy[0];
}

// ---------------------------------------------------------------------------
extern "C" void kernel_launch(void* const* d_in, const int* in_sizes, int n_in,
                              void* d_out, int out_size, void* d_ws, size_t ws_size,
                              hipStream_t stream) {
    const float* curr      = (const float*)d_in[0];
    const float* hist_xy   = (const float*)d_in[1];
    const float* hist_mask = (const float*)d_in[2];
    const float* fut       = (const float*)d_in[3];
    const float* ecW1 = (const float*)d_in[4];
    const float* ecb1 = (const float*)d_in[5];
    const float* ecW2 = (const float*)d_in[6];
    const float* ecb2 = (const float*)d_in[7];
    const float* efW1 = (const float*)d_in[8];
    const float* efb1 = (const float*)d_in[9];
    const float* efW2 = (const float*)d_in[10];
    const float* efb2 = (const float*)d_in[11];
    const float* eW1  = (const float*)d_in[12];
    const float* eb1  = (const float*)d_in[13];
    const float* eW2  = (const float*)d_in[14];
    const float* eb2  = (const float*)d_in[15];
    const float* eW3  = (const float*)d_in[16];
    const float* eb3  = (const float*)d_in[17];
    const float* dummy= (const float*)d_in[18];

    float* ws = (float*)d_ws;
    float* a     = ws;                          // NM*H floats
    float* b     = a + (size_t)NM * H;          // NN*NT*H floats
    float* xpred = b + (size_t)NN * NT * H;     // NN*NM*2 floats
    _Float16* bpack = (_Float16*)(xpred + (size_t)NN * NM * 2); // 16384 halfs (16B aligned)

    pack_w2_kernel<<<4, 256, 0, stream>>>(eW2, bpack);
    encode_kernel<<<2 * NM, 128, 0, stream>>>(curr, fut,
                                              ecW1, ecb1, ecW2, ecb2,
                                              efW1, efb1, efW2, efb2,
                                              eW1, a, b);
    xpred_kernel<<<(NM + 255) / 256, 256, 0, stream>>>(curr, hist_xy, hist_mask, xpred);
    assoc_fused_kernel<<<dim3(NT / 128, NM, NN), 256, 0, stream>>>(
        a, b, fut, xpred, bpack, eW1, eb1, eb2, eW3, eb3, dummy, (float*)d_out);
}
